// DOMGraphTransformer_214748365440
// MI455X (gfx1250) — compile-verified, matrix operand order checked by reading, not run on hardware
//
#include <hip/hip_runtime.h>
#include <hip/hip_bf16.h>
#include <math.h>

// ---------------------------------------------------------------------------
// Problem constants (from the reference)
// ---------------------------------------------------------------------------
#define NNODES 50000
#define NEDGES 400000
#define HEADS  4
#define LN_EPS 1e-5f
#define KPAD1  544     // 523 padded up to a multiple of 32

typedef __bf16 v16bf __attribute__((ext_vector_type(16)));
typedef float  v8f   __attribute__((ext_vector_type(8)));

union Frag {
    uint4 q[2];
    v16bf v;
};

__device__ __forceinline__ unsigned short f2bf(float f) {
    unsigned int u = __float_as_uint(f);
    u += 0x7fffu + ((u >> 16) & 1u);   // round-to-nearest-even bf16
    return (unsigned short)(u >> 16);
}

// ---------------------------------------------------------------------------
// One-time conversions: f32 -> zero-padded bf16 (row-major X, transposed W)
// ---------------------------------------------------------------------------
__global__ __launch_bounds__(256)
void cvt_x_bf16(const float* __restrict__ x, unsigned short* __restrict__ xb,
                int Nn, int K, int Kpad) {
    int t = blockIdx.x * blockDim.x + threadIdx.x;
    if (t >= Nn * Kpad) return;
    int n = t / Kpad, k = t - n * Kpad;
    float v = (k < K) ? x[(size_t)n * K + k] : 0.0f;
    xb[t] = f2bf(v);
}

__global__ __launch_bounds__(256)
void cvt_w_bf16_t(const float* __restrict__ W, unsigned short* __restrict__ Wt,
                  int K, int N, int Kpad) {
    // Wt[n][k] = W[k][n], zero-padded in k
    int t = blockIdx.x * blockDim.x + threadIdx.x;
    if (t >= N * Kpad) return;
    int n = t / Kpad, k = t - n * Kpad;
    float v = (k < K) ? W[(size_t)k * N + n] : 0.0f;
    Wt[t] = f2bf(v);
}

// ---------------------------------------------------------------------------
// GEMM: C[M x N] = A[M x Kpad](bf16) * Bt[N x Kpad](bf16,transposed), f32 acc.
// Block = 256 threads = 8 waves. Tile: 128(M) x 128(N), K-step 32.
// Double-buffered LDS fed by global_load_async_to_lds_b128 (ASYNCcnt);
// per K-step: batched ds_load_b128 clause -> 8 back-to-back WMMAs.
// ---------------------------------------------------------------------------
#define TM 128
#define TN 128
#define TKP 40   // 32 K-halves + 8 pad, keeps 16B alignment for ds_load_b128

__global__ __launch_bounds__(256)
void gat_gemm_bf16(const unsigned short* __restrict__ A,
                   const unsigned short* __restrict__ Bt,
                   float* __restrict__ C, int M, int Kpad, int N) {
    __shared__ __align__(16) unsigned short Asm[2][TM][TKP];
    __shared__ __align__(16) unsigned short Bsm[2][TN][TKP];  // [n][k]

    const int tid  = threadIdx.x;
    const int lane = tid & 31;
    const int wave = tid >> 5;
    const int m0   = blockIdx.x * TM;
    const int n0   = blockIdx.y * TN;
    const int mw   = wave * 16;

    // Staging assignment: thread -> (row, 16-element K half). Clamp the A row
    // (garbage lands only in accumulator rows the epilogue never stores).
    const int srow  = tid >> 1;          // 0..127
    const int shalf = (tid & 1) * 16;    // 0 or 16 (bf16 elements)
    const int gmr   = (m0 + srow < M) ? (m0 + srow) : (M - 1);
    const unsigned short* arow = A  + (size_t)gmr * Kpad + shalf;
    const unsigned short* brow = Bt + (size_t)(n0 + srow) * Kpad + shalf;

    // LDS destination byte offsets (low 32 bits of the shared aperture address)
    const unsigned aoff0 = (unsigned)(size_t)&Asm[0][srow][shalf];
    const unsigned aoff1 = (unsigned)(size_t)&Asm[1][srow][shalf];
    const unsigned boff0 = (unsigned)(size_t)&Bsm[0][srow][shalf];
    const unsigned boff1 = (unsigned)(size_t)&Bsm[1][srow][shalf];

    // Issue 4 async b128 copies (32B of A, 32B of B) into one LDS buffer.
    // The instruction offset applies to both global and LDS addresses.
    auto issue = [&](unsigned ao, unsigned bo, int k0) {
        unsigned long long ga = (unsigned long long)(arow + k0);
        unsigned long long gb = (unsigned long long)(brow + k0);
        asm volatile(
            "global_load_async_to_lds_b128 %0, %2, off\n\t"
            "global_load_async_to_lds_b128 %0, %2, off offset:16\n\t"
            "global_load_async_to_lds_b128 %1, %3, off\n\t"
            "global_load_async_to_lds_b128 %1, %3, off offset:16"
            :: "v"(ao), "v"(bo), "v"(ga), "v"(gb) : "memory");
    };

    v8f acc[8] = {};

    const int nk = Kpad >> 5;
    issue(aoff0, boff0, 0);

    for (int kt = 0; kt < nk; ++kt) {
        const int cur = kt & 1;
        if (kt + 1 < nk) {
            // Prefetch next tile into the other buffer, then wait until only
            // those 4 async ops remain in flight -> current buffer resident.
            issue(cur ? aoff0 : aoff1, cur ? boff0 : boff1, (kt + 1) << 5);
            asm volatile("s_wait_asynccnt 4" ::: "memory");
        } else {
            asm volatile("s_wait_asynccnt 0" ::: "memory");
        }
        __syncthreads();

        // Fetch A fragment + all 8 B fragments first (one LDS clause), then
        // run the 8 WMMAs back-to-back without per-op dscnt stalls.
        const int mrow = mw + (lane & 15);
        const int kh   = (lane >> 4) * 8;
        Frag af, bfr[8];
        af.q[0] = *(const uint4*)&Asm[cur][mrow][kh];        // K = kh..kh+7
        af.q[1] = *(const uint4*)&Asm[cur][mrow][kh + 16];   // K = 16+kh..
        #pragma unroll
        for (int s = 0; s < 8; ++s) {
            const int ncol = s * 16 + (lane & 15);
            bfr[s].q[0] = *(const uint4*)&Bsm[cur][ncol][kh];
            bfr[s].q[1] = *(const uint4*)&Bsm[cur][ncol][kh + 16];
        }
        #pragma unroll
        for (int s = 0; s < 8; ++s) {
            acc[s] = __builtin_amdgcn_wmma_f32_16x16x32_bf16(
                false, af.v, false, bfr[s].v, (short)0, acc[s], false, false);
        }
        __syncthreads();   // all waves done reading before buffer is re-filled
    }

    // Epilogue: VGPR r holds M = r + 8*(lane>=16); N always in range.
    #pragma unroll
    for (int s = 0; s < 8; ++s) {
        const int gn = n0 + s * 16 + (lane & 15);
        #pragma unroll
        for (int r = 0; r < 8; ++r) {
            int gm = m0 + mw + r + 8 * (lane >> 4);
            if (gm < M) C[(size_t)gm * N + gn] = acc[s][r];
        }
    }
}

// ---------------------------------------------------------------------------
// Per-node attention scores (float4-vectorized inner product)
// ---------------------------------------------------------------------------
__global__ __launch_bounds__(256)
void gat_node_scores(const float* __restrict__ Wx, const float* __restrict__ a,
                     float* __restrict__ ssrc, float* __restrict__ sdst,
                     int Nn, int dout, int hd) {
    int t = blockIdx.x * blockDim.x + threadIdx.x;
    if (t >= Nn * HEADS) return;
    int n = t >> 2, h = t & 3;
    const float4* row = (const float4*)(Wx + (size_t)n * dout + h * hd);
    const float4* a1  = (const float4*)(a);
    const float4* a2  = (const float4*)(a + hd);
    float s1 = 0.0f, s2 = 0.0f;
    const int q = hd >> 2;
    for (int j = 0; j < q; ++j) {
        float4 v = row[j];
        float4 p = a1[j];
        float4 r2 = a2[j];
        s1 += v.x * p.x + v.y * p.y + v.z * p.z + v.w * p.w;
        s2 += v.x * r2.x + v.y * r2.y + v.z * r2.z + v.w * r2.w;
    }
    ssrc[t] = s1;
    sdst[t] = s2;
}

// ---------------------------------------------------------------------------
// Edge pass 1: numerators + segment denominators (f32 global atomics)
// ---------------------------------------------------------------------------
__global__ __launch_bounds__(256)
void gat_edge_scores(const long long* __restrict__ src,
                     const long long* __restrict__ dst,
                     const float* __restrict__ ssrc, const float* __restrict__ sdst,
                     float* __restrict__ eexp, float* __restrict__ denom, int E_) {
    int t = blockIdx.x * blockDim.x + threadIdx.x;
    if (t >= E_ * HEADS) return;
    int e = t >> 2, h = t & 3;
    long long s = src[e], d = dst[e];
    float sc = ssrc[s * HEADS + h] + sdst[d * HEADS + h];
    sc = sc > 0.0f ? sc : 0.2f * sc;
    float ex = expf(sc);
    eexp[t] = ex;
    atomicAdd(&denom[d * HEADS + h], ex);
}

// ---------------------------------------------------------------------------
// Edge pass 2: agg[dst,h,:] += alpha * Wx[src,h,:]  (float4 loads, f32 atomics)
// ---------------------------------------------------------------------------
__global__ __launch_bounds__(256)
void gat_edge_aggregate(const long long* __restrict__ src,
                        const long long* __restrict__ dst,
                        const float* __restrict__ eexp,
                        const float* __restrict__ denom,
                        const float* __restrict__ Wx, float* __restrict__ agg,
                        int E_, int dout, int hd) {
    int t = blockIdx.x * blockDim.x + threadIdx.x;
    if (t >= E_ * HEADS) return;
    int e = t >> 2, h = t & 3;
    long long s = src[e], d = dst[e];
    float alpha = eexp[t] / (denom[d * HEADS + h] + 1e-8f);
    const float4* wrow = (const float4*)(Wx + (size_t)s * dout + h * hd);
    float* arow = agg + (size_t)d * dout + h * hd;
    const int q = hd >> 2;
    for (int j = 0; j < q; ++j) {
        float4 v = wrow[j];
        atomicAdd(&arow[j * 4 + 0], alpha * v.x);
        atomicAdd(&arow[j * 4 + 1], alpha * v.y);
        atomicAdd(&arow[j * 4 + 2], alpha * v.z);
        atomicAdd(&arow[j * 4 + 3], alpha * v.w);
    }
}

// ---------------------------------------------------------------------------
// ELU + LayerNorm, one wave32 per node. Writes f32 (final layer) and/or bf16
// (next layer's GEMM operand). dout in {256, 512}.
// ---------------------------------------------------------------------------
__global__ __launch_bounds__(256)
void gat_elu_ln(const float* __restrict__ agg, const float* __restrict__ g,
                const float* __restrict__ b, float* __restrict__ fout,
                unsigned short* __restrict__ bfout, int Nn, int dout) {
    int node = blockIdx.x * 8 + (threadIdx.x >> 5);
    int lane = threadIdx.x & 31;
    if (node >= Nn) return;
    const float* row = agg + (size_t)node * dout;
    const int cnt = dout >> 5;          // <= 16
    float vals[16];
    float sum = 0.0f, sq = 0.0f;
    for (int i = 0; i < cnt; ++i) {
        float v = row[lane + i * 32];
        v = v > 0.0f ? v : (expf(v) - 1.0f);   // ELU(alpha=1)
        vals[i] = v;
        sum += v;
        sq  += v * v;
    }
    #pragma unroll
    for (int m = 16; m >= 1; m >>= 1) {
        sum += __shfl_xor(sum, m, 32);
        sq  += __shfl_xor(sq,  m, 32);
    }
    float mean = sum / (float)dout;
    float var  = sq / (float)dout - mean * mean;
    float rstd = rsqrtf(var + LN_EPS);
    for (int i = 0; i < cnt; ++i) {
        int j = lane + i * 32;
        float o = (vals[i] - mean) * rstd * g[j] + b[j];
        if (fout)  fout[(size_t)node * dout + j] = o;
        if (bfout) bfout[(size_t)node * dout + j] = f2bf(o);
    }
}

// ---------------------------------------------------------------------------
// Host launcher
// ---------------------------------------------------------------------------
extern "C" void kernel_launch(void* const* d_in, const int* in_sizes, int n_in,
                              void* d_out, int out_size, void* d_ws, size_t ws_size,
                              hipStream_t stream) {
    (void)in_sizes; (void)n_in; (void)out_size; (void)ws_size;

    const int Nn = NNODES;
    const int E_ = NEDGES;
    static const int DIMS_[4] = {523, 256, 256, 512};

    const float*     x0  = (const float*)d_in[0];
    const long long* ei  = (const long long*)d_in[1];   // int64 edge_index [2,E]
    const long long* src = ei;
    const long long* dst = ei + E_;

    // Workspace layout (offsets 256B aligned)
    char* ws = (char*)d_ws;
    float*          Wx    = (float*)(ws + 0);            // N*512*4 = 102,400,000
    float*          agg   = (float*)(ws + 102400000);    // N*512*4
    unsigned short* Xbf   = (unsigned short*)(ws + 204800000); // N*544*2 = 54,400,000
    unsigned short* xAbf  = (unsigned short*)(ws + 259200000); // N*256*2 = 25,600,000
    unsigned short* xBbf  = (unsigned short*)(ws + 284800000); // N*256*2
    unsigned short* Wtbf  = (unsigned short*)(ws + 310400000); // <=512*544*2 ~ 557,056
    float*          ssrc  = (float*)(ws + 311040000);    // N*4*4 = 800,000
    float*          sdst  = (float*)(ws + 311840000);
    float*          denom = (float*)(ws + 312640000);
    float*          eexp  = (float*)(ws + 313440000);    // E*4*4 = 6,400,000

    // One-time: node_feats f32 -> padded bf16 (K 523 -> 544)
    {
        int tot = Nn * KPAD1;
        cvt_x_bf16<<<(tot + 255) / 256, 256, 0, stream>>>(x0, Xbf, Nn, 523, KPAD1);
    }

    const unsigned short* xb = Xbf;
    int Kpad = KPAD1;

    unsigned short* nxtbf[3] = {xAbf, xBbf, nullptr};

    for (int li = 0; li < 3; ++li) {
        const int din  = DIMS_[li];
        const int dout = DIMS_[li + 1];
        const int hd   = dout / HEADS;
        const float* W  = (const float*)d_in[2 + li * 4];
        const float* a  = (const float*)d_in[3 + li * 4];
        const float* g  = (const float*)d_in[4 + li * 4];
        const float* bb = (const float*)d_in[5 + li * 4];

        // 0) W f32 -> transposed padded bf16
        {
            int tot = dout * Kpad;
            cvt_w_bf16_t<<<(tot + 255) / 256, 256, 0, stream>>>(W, Wtbf, din, dout, Kpad);
        }

        // 1) Wx = x @ W  (bf16 WMMA, f32 accumulate; async-LDS double buffer)
        dim3 gg((Nn + TM - 1) / TM, dout / TN);
        gat_gemm_bf16<<<gg, 256, 0, stream>>>(xb, Wtbf, Wx, Nn, Kpad, dout);

        // 2) Per-node attention score halves
        gat_node_scores<<<(Nn * HEADS + 255) / 256, 256, 0, stream>>>(
            Wx, a, ssrc, sdst, Nn, dout, hd);

        // 3) Edge softmax numerators + segment denominators
        hipMemsetAsync(denom, 0, (size_t)Nn * HEADS * sizeof(float), stream);
        gat_edge_scores<<<(E_ * HEADS + 255) / 256, 256, 0, stream>>>(
            src, dst, ssrc, sdst, eexp, denom, E_);

        // 4) Weighted scatter-add aggregation
        hipMemsetAsync(agg, 0, (size_t)Nn * dout * sizeof(float), stream);
        gat_edge_aggregate<<<(E_ * HEADS + 255) / 256, 256, 0, stream>>>(
            src, dst, eexp, denom, Wx, agg, E_, dout, hd);

        // 5) ELU + LayerNorm -> next layer bf16 operand (or final f32 output)
        float* fout = (li == 2) ? (float*)d_out : nullptr;
        gat_elu_ln<<<(Nn + 7) / 8, 256, 0, stream>>>(
            agg, g, bb, fout, nxtbf[li], Nn, dout);

        xb = nxtbf[li];
        Kpad = dout;   // 256, already a multiple of 32
    }
}
